// PHVGNNModel_15049565405195
// MI455X (gfx1250) — compile-verified, matrix operand order checked by reading, not run on hardware
//
#include <hip/hip_runtime.h>
#include <hip/hip_bf16.h>
#include <hip/hip_fp16.h>

#define NSEQ  4096
#define TSEQ  128
#define WEMB  64
#define LHID  128
#define HIDF  128
#define NEDGE 131072

typedef __attribute__((ext_vector_type(16))) _Float16 v16h;
typedef __attribute__((ext_vector_type(8)))  _Float16 v8h;
typedef __attribute__((ext_vector_type(8)))  float    v8f;

// ---------------------------------------------------------------------------
// WMMA fragment helpers (gfx1250 wave32, CDNA5 ISA):
// 16-bit A/B fragment: lane L<16 holds row L,   K = {k0..k0+7, k0+16..k0+23};
//                      lane L>=16 holds row L-16, K = {k0+8..k0+15, k0+24..k0+31}.
// ---------------------------------------------------------------------------
__device__ __forceinline__ v16h make_frag(const _Float16* p) {
  v8h lo = *(const v8h*)(p);
  v8h hi = *(const v8h*)(p + 16);
  return __builtin_shufflevector(lo, hi, 0,1,2,3,4,5,6,7,8,9,10,11,12,13,14,15);
}

// Swizzled (fragment-contiguous) storage: fragment f occupies 1KB; lane l's
// 16 halves (32B) are contiguous at f*512 + l*16 halves -> coalesced loads,
// and all hot-loop accesses are base + compile-time immediate offsets.
__device__ __forceinline__ v16h load_frag32(const _Float16* p) {
  v8h lo = *(const v8h*)(p);
  v8h hi = *(const v8h*)(p + 8);
  return __builtin_shufflevector(lo, hi, 0,1,2,3,4,5,6,7,8,9,10,11,12,13,14,15);
}

// Map swizzled flat element index -> (row, k) of a row-major (Nrows x K) matrix.
__device__ __forceinline__ void sw_decomp(int i, int K, int& row, int& k) {
  int e    = i & 15;
  int lane = (i >> 4) & 31;
  int rest = i >> 9;                 // tile*nkk + kk
  int nkk  = K >> 5;
  int kk   = rest % nkk;
  int tile = rest / nkk;
  row = tile * 16 + (lane & 15);
  int ko = (lane & 16) ? 8 : 0;
  k = kk * 32 + ko + (e < 8 ? e : e + 8);
}

__device__ __forceinline__ float sigm(float x) { return 1.0f / (1.0f + __expf(-x)); }
__device__ __forceinline__ float tanh_fast(float x) {
  float e = __expf(2.0f * x);
  return (e - 1.0f) / (e + 1.0f);
}

// ---------------------------------------------------------------------------
// f32 -> f16 swizzle pack (weights once per launch; also activations re-layout)
// ---------------------------------------------------------------------------
__global__ void __launch_bounds__(256)
swizzle_w_kernel(const float* __restrict__ W, _Float16* __restrict__ out, int Nrows, int K) {
  int i = blockIdx.x * 256 + threadIdx.x;
  if (i >= Nrows * K) return;
  int row, k; sw_decomp(i, K, row, k);
  out[i] = (_Float16)W[(size_t)row * K + k];
}

// ---------------------------------------------------------------------------
// Y = X(MxK) @ W(NxK)^T + bias ; both operands swizzled. K templated so the
// k-loop fully unrolls into immediate-offset b128 clauses.
// 256 thr / 8 waves, 64x64 tile. Wave w: m-subtile w>>1, n-tile pair w&1.
// ---------------------------------------------------------------------------
template <int K>
__global__ void __launch_bounds__(256)
gemm_sw_kernel(const _Float16* __restrict__ Xsw, const _Float16* __restrict__ Wsw,
               const float* __restrict__ bias, float* __restrict__ Y,
               int N, int relu)
{
  constexpr int nkk = K >> 5;
  const int lane   = threadIdx.x & 31;
  const int w      = threadIdx.x >> 5;
  const int atile  = blockIdx.x * 4 + (w >> 1);
  const int btile0 = blockIdx.y * 4 + (w & 1) * 2;

  const _Float16* ab = Xsw + (size_t)atile  * (nkk * 512) + lane * 16;
  const _Float16* bb = Wsw + (size_t)btile0 * (nkk * 512) + lane * 16;

  v8f acc0 = {}; v8f acc1 = {};
#pragma unroll
  for (int kk = 0; kk < nkk; ++kk) {
    __builtin_prefetch(ab + (kk + 2) * 512, 0, 0);
    v16h a  = load_frag32(ab + kk * 512);
    v16h b0 = load_frag32(bb + kk * 512);
    v16h b1 = load_frag32(bb + (nkk + kk) * 512);
    acc0 = __builtin_amdgcn_wmma_f32_16x16x32_f16(false, a, false, b0, (short)0, acc0, false, false);
    acc1 = __builtin_amdgcn_wmma_f32_16x16x32_f16(false, a, false, b1, (short)0, acc1, false, false);
  }

  const int r15 = lane & 15;
  const int mb  = atile * 16 + ((lane & 16) ? 8 : 0);
  const int c0  = btile0 * 16 + r15, c1 = c0 + 16;
  const float b0v = bias ? bias[c0] : 0.0f;
  const float b1v = bias ? bias[c1] : 0.0f;
#pragma unroll
  for (int r = 0; r < 8; ++r) {
    float v0 = acc0[r] + b0v, v1 = acc1[r] + b1v;
    if (relu) { v0 = fmaxf(v0, 0.0f); v1 = fmaxf(v1, 0.0f); }
    Y[(size_t)(mb + r) * N + c0] = v0;
    Y[(size_t)(mb + r) * N + c1] = v1;
  }
}

// ---------------------------------------------------------------------------
// Fused bidirectional-capable LSTM scan: one launch covers all T steps.
// Block owns 32 sequences; h state in LDS (f16, padded rows -> conflict-free
// ds_load_b128 fragments), c state in registers. Loop-invariant weights are
// staged in LDS (Whh always: 128KB; Wih too when DIN==64: +64KB); layer-1/2
// Wih (256KB) streams from L2. An iteration memory barrier stops LICM from
// hoisting the unrolled weight fragments into registers (which spilled).
// Wave w: m-subtile (w&1), hidden cols [(w>>1)*32, +32) for ALL 4 gates.
// Backward handled by per-lane rev-index loads/stores (no flip buffers).
// ---------------------------------------------------------------------------
#define HPAD (LHID + 8)
template <int DIN>
__global__ void __launch_bounds__(256)
lstm_scan_kernel(const _Float16* __restrict__ in,
                 const _Float16* __restrict__ WihSw,  // swizzled 512 x DIN
                 const _Float16* __restrict__ WhhSw,  // swizzled 512 x 128
                 const float* __restrict__ bias,      // 512 = bih + bhh
                 const int* __restrict__ seq_len,
                 _Float16* __restrict__ out,          // (N*T) x 256, cols [col0,col0+128)
                 int col0, int backward)
{
  constexpr int  nkkI   = DIN >> 5;
  constexpr bool IH_LDS = (DIN == 64);
  __shared__ _Float16 h_sh[32][HPAD];
  __shared__ int Lsh[32];
  __shared__ _Float16 whh_l[512 * 128];
  __shared__ _Float16 wih_l[IH_LDS ? 512 * 64 : 8];

  const int lane  = threadIdx.x & 31;
  const int w     = threadIdx.x >> 5;
  const int rbase = (w & 1) * 16;
  const int hw    = w >> 1;            // 0..3 -> hidden cols hw*32..+32
  const int n0    = blockIdx.x * 32;
  const int r15   = lane & 15;
  const int ko    = (lane & 16) ? 8 : 0;

  // stage weights into LDS (fragment layout preserved; coalesced 16B copies)
  for (int i = threadIdx.x * 8; i < 512 * 128; i += 256 * 8)
    *(v8h*)&whh_l[i] = *(const v8h*)&WhhSw[i];
  if constexpr (IH_LDS) {
    for (int i = threadIdx.x * 8; i < 512 * DIN; i += 256 * 8)
      *(v8h*)&wih_l[i] = *(const v8h*)&WihSw[i];
  }
  for (int i = threadIdx.x; i < 32 * HPAD; i += 256) ((_Float16*)h_sh)[i] = (_Float16)0.0f;
  if (threadIdx.x < 32) Lsh[threadIdx.x] = seq_len[n0 + threadIdx.x];
  __syncthreads();

  const int arow = rbase + r15;          // A-fragment row this lane serves
  const int La   = Lsh[arow];

  const int ibase = hw * (2 * nkkI * 512) + lane * 16;   // Wih fragment base (halves)
  const int hbase = hw * (2 * 4 * 512)    + lane * 16;   // Whh fragment base (halves)

  float bcol[4][2];
#pragma unroll
  for (int g = 0; g < 4; ++g)
#pragma unroll
    for (int j = 0; j < 2; ++j)
      bcol[g][j] = bias[(g * 8 + hw * 2 + j) * 16 + r15];

  const int erow0 = rbase + ((lane & 16) ? 8 : 0);
  int Lrow[8];
  unsigned obase[2][8];                  // 32-bit store offsets (sans p*256)
#pragma unroll
  for (int r = 0; r < 8; ++r) Lrow[r] = Lsh[erow0 + r];
#pragma unroll
  for (int j = 0; j < 2; ++j)
#pragma unroll
    for (int r = 0; r < 8; ++r)
      obase[j][r] = (unsigned)(n0 + erow0 + r) * (TSEQ * 2 * LHID)
                  + (unsigned)(col0 + hw * 32 + j * 16 + r15);

  float c[2][8];
#pragma unroll
  for (int j = 0; j < 2; ++j)
#pragma unroll
    for (int r = 0; r < 8; ++r) c[j][r] = 0.0f;

  const unsigned xrowbase = (unsigned)(n0 + arow) * (TSEQ * DIN);

  for (int s = 0; s < TSEQ; ++s) {
    asm volatile("" ::: "memory");       // keep per-step loads inside the loop
    const int tin = backward ? (s < La ? La - 1 - s : s) : s;
    const unsigned xoff = xrowbase + (unsigned)tin * DIN + (unsigned)ko;
    // hint next timestep's activation row into cache
    const int s2   = (s + 1 < TSEQ) ? s + 1 : s;
    const int tin2 = backward ? (s2 < La ? La - 1 - s2 : s2) : s2;
    __builtin_prefetch(in + xrowbase + (size_t)tin2 * DIN + ko, 0, 0);

    v8f acc[4][2];
#pragma unroll
    for (int g = 0; g < 4; ++g)
#pragma unroll
      for (int j = 0; j < 2; ++j) {
        v8f t;
#pragma unroll
        for (int r = 0; r < 8; ++r) t[r] = bcol[g][j];
        acc[g][j] = t;
      }

    // input contribution: x_t @ Wih^T
#pragma unroll
    for (int kki = 0; kki < nkkI; ++kki) {
      v16h a = make_frag(in + xoff + kki * 32);
#pragma unroll
      for (int g = 0; g < 4; ++g)
#pragma unroll
        for (int j = 0; j < 2; ++j) {
          v16h b;
          if constexpr (IH_LDS)
            b = load_frag32(&wih_l[ibase + ((g * 8 + j) * nkkI + kki) * 512]);
          else
            b = load_frag32(WihSw + ibase + ((g * 8 + j) * nkkI + kki) * 512);
          acc[g][j] = __builtin_amdgcn_wmma_f32_16x16x32_f16(false, a, false, b, (short)0, acc[g][j], false, false);
        }
    }
    // recurrent contribution: h @ Whh^T (A + B both from LDS)
#pragma unroll
    for (int kki = 0; kki < 4; ++kki) {
      v16h a = make_frag(&h_sh[arow][kki * 32 + ko]);
#pragma unroll
      for (int g = 0; g < 4; ++g)
#pragma unroll
        for (int j = 0; j < 2; ++j) {
          v16h b = load_frag32(&whh_l[hbase + ((g * 8 + j) * 4 + kki) * 512]);
          acc[g][j] = __builtin_amdgcn_wmma_f32_16x16x32_f16(false, a, false, b, (short)0, acc[g][j], false, false);
        }
    }
    __syncthreads();   // all h_sh reads done before overwrite

    // cell update: all 4 gates for this wave's (rows, hidden cols) in registers
#pragma unroll
    for (int j = 0; j < 2; ++j) {
      const int col = hw * 32 + j * 16 + r15;
#pragma unroll
      for (int r = 0; r < 8; ++r) {
        float iv = acc[0][j][r], fv = acc[1][j][r];
        float gv = acc[2][j][r], ov = acc[3][j][r];
        float cn = sigm(fv) * c[j][r] + sigm(iv) * tanh_fast(gv);
        float hv = sigm(ov) * tanh_fast(cn);
        c[j][r] = cn;
        const int row = erow0 + r;
        h_sh[row][col] = (_Float16)hv;               // raw state for next step
        const int L = Lrow[r];
        const int p = backward ? (s < L ? L - 1 - s : s) : s;
        const float val = (p < L) ? hv : 0.0f;       // mask only the output
        out[obase[j][r] + ((unsigned)p << 8)] = (_Float16)val;
      }
    }
    __syncthreads();   // h_sh writes visible before next step's reads
  }
}

// ------------------------- small elementwise kernels -----------------------
__global__ void __launch_bounds__(256) bias_sum_kernel(const float* a, const float* b, float* out, int n) {
  int i = blockIdx.x * 256 + threadIdx.x;
  if (i < n) out[i] = a[i] + b[i];
}
__global__ void __launch_bounds__(256) zero_f32_kernel(float* p, int n) {
  int i = blockIdx.x * 256 + threadIdx.x;
  if (i < n) p[i] = 0.0f;
}
__global__ void __launch_bounds__(256) embed_kernel(const int* __restrict__ x,
                                                    const float* __restrict__ emb,
                                                    _Float16* __restrict__ out, int total) {
  int i = blockIdx.x * 256 + threadIdx.x;
  if (i < total) out[i] = (_Float16)emb[(size_t)x[i >> 6] * WEMB + (i & 63)];
}
// gather last-timestep rows directly into swizzled-A layout for the lin GEMM
__global__ void __launch_bounds__(256) last_gather_sw_kernel(const _Float16* __restrict__ buf,
                                                             const int* __restrict__ seq_len,
                                                             _Float16* __restrict__ out, int total) {
  int i = blockIdx.x * 256 + threadIdx.x;
  if (i >= total) return;
  int n, k; sw_decomp(i, 256, n, k);
  int L = seq_len[n];
  out[i] = buf[((size_t)n * TSEQ + (L - 1)) * 256 + k];
}
__global__ void __launch_bounds__(256) deg_kernel(const int* __restrict__ ei, float* __restrict__ deg, int ne) {
  int i = blockIdx.x * 256 + threadIdx.x;
  if (i < ne) atomicAdd(&deg[ei[ne + i]], 1.0f);
}
__global__ void __launch_bounds__(256) dis_kernel(const float* __restrict__ deg, float* __restrict__ dis, int n) {
  int i = blockIdx.x * 256 + threadIdx.x;
  if (i < n) dis[i] = rsqrtf(deg[i] + 1.0f);   // +1 = self loop; always > 0
}
__global__ void __launch_bounds__(256) gcn_scatter_kernel(const int* __restrict__ ei,
                                                          const float* __restrict__ xw,
                                                          const float* __restrict__ dis,
                                                          float* __restrict__ agg, int ne) {
  long long i = (long long)blockIdx.x * 256 + threadIdx.x;
  if (i >= (long long)ne * HIDF) return;
  int e = (int)(i >> 7), k = (int)(i & 127);
  int row = ei[e], col = ei[ne + e];
  float nw = dis[row] * dis[col];
  atomicAdd(&agg[(size_t)col * HIDF + k], xw[(size_t)row * HIDF + k] * nw);
}
__global__ void __launch_bounds__(256) gcn_self_kernel(const float* __restrict__ xw,
                                                       const float* __restrict__ dis,
                                                       float* __restrict__ agg, int total) {
  int i = blockIdx.x * 256 + threadIdx.x;
  if (i >= total) return;
  int n = i >> 7;
  agg[i] += xw[i] * dis[n] * dis[n];
}
// GCN epilogue -> swizzled-A layout (feeds next GEMM directly)
__global__ void __launch_bounds__(256) bias_relu_sw_kernel(const float* __restrict__ agg,
                                                           const float* __restrict__ b,
                                                           _Float16* __restrict__ hsw, int total) {
  int i = blockIdx.x * 256 + threadIdx.x;
  if (i >= total) return;
  int n, k; sw_decomp(i, 128, n, k);
  hsw[i] = (_Float16)fmaxf(agg[(size_t)n * 128 + k] + b[k], 0.0f);
}
// GCN epilogue -> linear layout (h2 is consumed by per-node edge gather)
__global__ void __launch_bounds__(256) bias_relu_lin_kernel(const float* __restrict__ agg,
                                                            const float* __restrict__ b,
                                                            _Float16* __restrict__ h16, int total) {
  int i = blockIdx.x * 256 + threadIdx.x;
  if (i >= total) return;
  h16[i] = (_Float16)fmaxf(agg[i] + b[i & 127], 0.0f);
}
// edge-feature gather directly into swizzled-A layout for the o1 GEMM
__global__ void __launch_bounds__(256) z_gather_sw_kernel(const int* __restrict__ pos,
                                                          const int* __restrict__ neg,
                                                          const _Float16* __restrict__ h2,
                                                          _Float16* __restrict__ zsw, int total) {
  int i = blockIdx.x * 256 + threadIdx.x;
  if (i >= total) return;
  int e, k; sw_decomp(i, 256, e, k);
  int a, b;
  if (e < NEDGE) { a = pos[e]; b = pos[NEDGE + e]; }
  else           { a = neg[e - NEDGE]; b = neg[e - NEDGE + NEDGE]; }
  int src = (k < 128) ? a : b;
  zsw[i] = h2[(size_t)src * 128 + (k & 127)];
}
__global__ void __launch_bounds__(256) o2_kernel(const float* __restrict__ a1,
                                                 const float* __restrict__ o2w,
                                                 const float* __restrict__ o2b,
                                                 float* __restrict__ out, int ne2) {
  int e = blockIdx.x * 256 + threadIdx.x;
  if (e >= ne2) return;
  const float* ar = a1 + (size_t)e * 128;
  float s0 = o2b[0], s1 = o2b[1];
#pragma unroll 4
  for (int k = 0; k < 128; ++k) { s0 += ar[k] * o2w[k]; s1 += ar[k] * o2w[128 + k]; }
  out[(size_t)e * 2 + 0] = s0;
  out[(size_t)e * 2 + 1] = s1;
}

// ---------------------------------------------------------------------------
extern "C" void kernel_launch(void* const* d_in, const int* in_sizes, int n_in,
                              void* d_out, int out_size, void* d_ws, size_t ws_size,
                              hipStream_t stream) {
  (void)in_sizes; (void)n_in; (void)out_size; (void)ws_size;
  const int NT = NSEQ * TSEQ;
  const int E2 = 2 * NEDGE;

  const int*   x       = (const int*)d_in[0];
  const int*   pos_ei  = (const int*)d_in[1];
  const int*   neg_ei  = (const int*)d_in[2];
  const int*   seq_len = (const int*)d_in[3];
  const float* emb     = (const float*)d_in[4];
  const float* lw[3][8];
  for (int l = 0; l < 3; ++l)
    for (int i = 0; i < 8; ++i) lw[l][i] = (const float*)d_in[5 + l * 8 + i];
  const float* lin_w = (const float*)d_in[29]; const float* lin_b = (const float*)d_in[30];
  const float* g1_w  = (const float*)d_in[31]; const float* g1_b  = (const float*)d_in[32];
  const float* g2_w  = (const float*)d_in[33]; const float* g2_b  = (const float*)d_in[34];
  const float* o1_w  = (const float*)d_in[35]; const float* o1_b  = (const float*)d_in[36];
  const float* o2_w  = (const float*)d_in[37]; const float* o2_b  = (const float*)d_in[38];
  float* out = (float*)d_out;

  // workspace bump allocator
  char* base = (char*)d_ws; size_t off = 0;
  auto alloc = [&](size_t bytes) -> void* {
    void* p = base + off; off += (bytes + 255) & ~(size_t)255; return p;
  };
  _Float16* buf0 = (_Float16*)alloc((size_t)NT * 64 * 2);
  _Float16* bufA = (_Float16*)alloc((size_t)NT * 256 * 2);
  _Float16* bufB = (_Float16*)alloc((size_t)NT * 256 * 2);
  _Float16* wihsw[3][2]; _Float16* whhsw[3][2]; float* bsum[3][2];
  for (int l = 0; l < 3; ++l) {
    int din = (l == 0) ? 64 : 256;
    for (int d = 0; d < 2; ++d) {
      wihsw[l][d] = (_Float16*)alloc((size_t)512 * din * 2);
      whhsw[l][d] = (_Float16*)alloc((size_t)512 * 128 * 2);
      bsum[l][d]  = (float*)alloc(512 * 4);
    }
  }
  _Float16* linwsw = (_Float16*)alloc(128 * 256 * 2);
  _Float16* g1wsw  = (_Float16*)alloc(128 * 128 * 2);
  _Float16* g2wsw  = (_Float16*)alloc(128 * 128 * 2);
  _Float16* o1wsw  = (_Float16*)alloc(128 * 256 * 2);
  _Float16* lastsw = (_Float16*)alloc((size_t)NSEQ * 256 * 2);
  float*    feat32 = (float*)alloc((size_t)NSEQ * 128 * 4);
  _Float16* featsw = (_Float16*)alloc((size_t)NSEQ * 128 * 2);
  float*    xw     = (float*)alloc((size_t)NSEQ * 128 * 4);
  float*    deg    = (float*)alloc((size_t)NSEQ * 4);
  float*    dis    = (float*)alloc((size_t)NSEQ * 4);
  float*    agg    = (float*)alloc((size_t)NSEQ * 128 * 4);
  _Float16* h1sw   = (_Float16*)alloc((size_t)NSEQ * 128 * 2);
  _Float16* h2_16  = (_Float16*)alloc((size_t)NSEQ * 128 * 2);
  _Float16* zsw    = (_Float16*)alloc((size_t)E2 * 256 * 2);
  float*    a1     = (float*)alloc((size_t)E2 * 128 * 4);

  // --- weight swizzle (f32 -> f16 fragment-contiguous), once per launch ---
  for (int l = 0; l < 3; ++l) {
    int din = (l == 0) ? 64 : 256;
    for (int d = 0; d < 2; ++d) {
      int nih = 512 * din;
      swizzle_w_kernel<<<(nih + 255) / 256, 256, 0, stream>>>(lw[l][d * 4 + 0], wihsw[l][d], 512, din);
      swizzle_w_kernel<<<(512 * 128 + 255) / 256, 256, 0, stream>>>(lw[l][d * 4 + 1], whhsw[l][d], 512, 128);
      bias_sum_kernel<<<2, 256, 0, stream>>>(lw[l][d * 4 + 2], lw[l][d * 4 + 3], bsum[l][d], 512);
    }
  }
  swizzle_w_kernel<<<(128 * 256 + 255) / 256, 256, 0, stream>>>(lin_w, linwsw, 128, 256);
  swizzle_w_kernel<<<(128 * 128 + 255) / 256, 256, 0, stream>>>(g1_w, g1wsw, 128, 128);
  swizzle_w_kernel<<<(128 * 128 + 255) / 256, 256, 0, stream>>>(g2_w, g2wsw, 128, 128);
  swizzle_w_kernel<<<(128 * 256 + 255) / 256, 256, 0, stream>>>(o1_w, o1wsw, 128, 256);

  // --- embedding gather -> f16 ---
  embed_kernel<<<(NT * 64) / 256, 256, 0, stream>>>(x, emb, buf0, NT * 64);

  // --- 3-layer bidirectional LSTM, fully fused scans ---
  _Float16* louts[3] = { bufA, bufB, bufA };
  lstm_scan_kernel<64><<<NSEQ / 32, 256, 0, stream>>>(buf0, wihsw[0][0], whhsw[0][0],
                                                      bsum[0][0], seq_len, louts[0], 0, 0);
  lstm_scan_kernel<64><<<NSEQ / 32, 256, 0, stream>>>(buf0, wihsw[0][1], whhsw[0][1],
                                                      bsum[0][1], seq_len, louts[0], 128, 1);
  const _Float16* lin_in = louts[0];
  for (int l = 1; l < 3; ++l) {
    _Float16* lout = louts[l];
    lstm_scan_kernel<256><<<NSEQ / 32, 256, 0, stream>>>(lin_in, wihsw[l][0], whhsw[l][0],
                                                         bsum[l][0], seq_len, lout, 0, 0);
    lstm_scan_kernel<256><<<NSEQ / 32, 256, 0, stream>>>(lin_in, wihsw[l][1], whhsw[l][1],
                                                         bsum[l][1], seq_len, lout, 128, 1);
    lin_in = lout;
  }
  const _Float16* h3 = bufA;

  // --- last-timestep gather (swizzled) + lin projection ---
  last_gather_sw_kernel<<<(NSEQ * 256) / 256, 256, 0, stream>>>(h3, seq_len, lastsw, NSEQ * 256);
  gemm_sw_kernel<256><<<dim3(NSEQ / 64, 2), 256, 0, stream>>>(lastsw, linwsw, lin_b, feat32, 128, 0);
  swizzle_w_kernel<<<(NSEQ * 128) / 256, 256, 0, stream>>>(feat32, featsw, NSEQ, 128);

  // --- two GCN layers ---
  auto run_gcn = [&](const _Float16* finsw, const int* ei, const _Float16* wsw,
                     const float* bvec, _Float16* hout, int sw_out) {
    zero_f32_kernel<<<(NSEQ + 255) / 256, 256, 0, stream>>>(deg, NSEQ);
    zero_f32_kernel<<<(NSEQ * 128) / 256, 256, 0, stream>>>(agg, NSEQ * 128);
    deg_kernel<<<(NEDGE + 255) / 256, 256, 0, stream>>>(ei, deg, NEDGE);
    dis_kernel<<<(NSEQ + 255) / 256, 256, 0, stream>>>(deg, dis, NSEQ);
    gemm_sw_kernel<128><<<dim3(NSEQ / 64, 2), 256, 0, stream>>>(finsw, wsw, nullptr, xw, 128, 0);
    gcn_scatter_kernel<<<((long long)NEDGE * 128) / 256, 256, 0, stream>>>(ei, xw, dis, agg, NEDGE);
    gcn_self_kernel<<<(NSEQ * 128) / 256, 256, 0, stream>>>(xw, dis, agg, NSEQ * 128);
    if (sw_out)
      bias_relu_sw_kernel<<<(NSEQ * 128) / 256, 256, 0, stream>>>(agg, bvec, hout, NSEQ * 128);
    else
      bias_relu_lin_kernel<<<(NSEQ * 128) / 256, 256, 0, stream>>>(agg, bvec, hout, NSEQ * 128);
  };
  run_gcn(featsw, pos_ei, g1wsw, g1_b, h1sw, 1);
  run_gcn(h1sw,   neg_ei, g2wsw, g2_b, h2_16, 0);

  // --- edge MLP ---
  z_gather_sw_kernel<<<((size_t)E2 * 256) / 256, 256, 0, stream>>>(pos_ei, neg_ei, h2_16, zsw,
                                                                   E2 * 256);
  gemm_sw_kernel<256><<<dim3(E2 / 64, 2), 256, 0, stream>>>(zsw, o1wsw, o1_b, a1, 128, 1);
  o2_kernel<<<(E2 + 255) / 256, 256, 0, stream>>>(a1, o2_w, o2_b, out, E2);
}
